// gn_attn_59828894433529
// MI455X (gfx1250) — matrix operand order repaired and checked by reading, not compiled, not run on hardware
//
#include <hip/hip_runtime.h>
#include <math.h>

// Problem constants (from the reference): B=4, S=4096, D=256, H=8, HD=32, E=262144
#define NNODE 16384      // B*S
#define DIM   256
#define NH    8
#define HD_   32
#define NE    262144
#define MQKV  768        // 3*D

typedef __attribute__((ext_vector_type(16))) __bf16 v16bf;
typedef __attribute__((ext_vector_type(8)))  float  v8f;
typedef __attribute__((ext_vector_type(4)))  int    v4i;

typedef __attribute__((address_space(1))) v4i* gv4i_p;   // global
typedef __attribute__((address_space(3))) v4i* lv4i_p;   // LDS

#if __has_builtin(__builtin_amdgcn_global_load_async_to_lds_b128) && \
    __has_builtin(__builtin_amdgcn_s_wait_asynccnt)
#define HAVE_ASYNC_LDS 1
#endif

// ---- helpers -------------------------------------------------------------

// order-preserving float<->uint transform for atomicMax on floats
__device__ __forceinline__ unsigned f_orderable(float f) {
  unsigned u = __float_as_uint(f);
  return (u & 0x80000000u) ? ~u : (u | 0x80000000u);
}
__device__ __forceinline__ float f_unorderable(unsigned k) {
  unsigned u = (k & 0x80000000u) ? (k & 0x7FFFFFFFu) : ~k;
  return __uint_as_float(u);
}

// ---- WMMA GEMM: C[N x MCOLS] = A[N x 256] * B[256 x MCOLS] + bias (+epilogue)
// Block = 8 waves = one 16-row tile x 128 cols.  A tile (16x256 f32, 16 KB) is
// staged in LDS once per block (async-to-LDS when available) and shared by all
// 8 waves; each wave owns one 16x16 output tile and runs 8 unrolled
// v_wmma_f32_16x16x32_bf16 ops (fp32 accumulate).  MCOLS is compile-time so
// all strided B/C offsets become instruction-immediate offsets.
// MODE 0: qkv projection, scale cols < DIM (the Q block) by qscale
// MODE 1: out projection, add residual[row*DIM + col]
template <int MODE, int MCOLS>
__global__ __launch_bounds__(256)
void k_wmma_gemm(const float* __restrict__ A, const float* __restrict__ B,
                 const float* __restrict__ bias, const float* __restrict__ residual,
                 float* __restrict__ C, float qscale)
{
  constexpr int K = DIM;
  __shared__ float As[16 * K];                    // 16 KB f32 A tile

  const int lane = threadIdx.x & 31;
  const int wid  = threadIdx.x >> 5;
  constexpr int colBlocks = MCOLS >> 7;           // 128 cols per block
  const int tn = blockIdx.x / colBlocks;          // row-tile (16 rows)
  const int cb = blockIdx.x % colBlocks;
  const int tm = cb * 8 + wid;                    // this wave's 16-col tile

  // ---- stage A tile: 4096 floats = 1024 float4 chunks, 4 per thread ------
  #pragma unroll
  for (int j = 0; j < 4; ++j) {
    int c    = threadIdx.x + 256 * j;             // chunk id
    int row  = c >> 6;                            // 64 float4 per row
    int col4 = (c & 63) << 2;
    const float* gp = A + (size_t)(tn * 16 + row) * K + col4;
    float*       lp = &As[row * K + col4];
#ifdef HAVE_ASYNC_LDS
    __builtin_amdgcn_global_load_async_to_lds_b128(
        (gv4i_p)(v4i*)(void*)const_cast<float*>(gp),
        (lv4i_p)(v4i*)(void*)lp, 0, 0);
#else
    *(float4*)lp = *(const float4*)gp;
#endif
  }
#ifdef HAVE_ASYNC_LDS
  __builtin_amdgcn_s_wait_asynccnt(0);
#endif
  __syncthreads();

  const int hi = lane >> 4;                       // lane half
  const int r  = lane & 15;
  const int col = tm * 16 + r;
  const float* bp = B + col;                      // B column for this lane
  __builtin_prefetch(bp, 0, 0);                   // global_prefetch_b8

  v8f acc = {};
  #pragma unroll
  for (int kk = 0; kk < K; kk += 32) {
    v16bf af, bfr;
    #pragma unroll
    for (int i = 0; i < 8; ++i) {
      // A 16x32 bf16 layout: lanes 0-15 -> K {0..7,16..23}, lanes 16-31 -> +8
      int ka = kk + 2 * i + ((i & 4) << 1) + hi * 8;
      float2 t = *(const float2*)&As[r * K + ka];
      af[2 * i]     = (__bf16)t.x;                // native RNE f32->bf16
      af[2 * i + 1] = (__bf16)t.y;
      // B 32x16 bf16 layout: lane = column; lanes 0-15 K {0..7,16..23}, lanes 16-31 -> +8
      int kb = kk + 2 * i + ((i & 4) << 1) + hi * 8;
      bfr[2 * i]     = (__bf16)bp[kb * MCOLS];        // compile-time strides ->
      bfr[2 * i + 1] = (__bf16)bp[(kb + 1) * MCOLS];  // immediate-offset loads
    }
    acc = __builtin_amdgcn_wmma_f32_16x16x32_bf16(
        false, af, false, bfr, (short)0, acc, false, false);
  }

  // C/D layout: VGPR rr -> M = rr + 8*hi, N = lane&15
  const float cscale = (MODE == 0) ? ((col < DIM) ? qscale : 1.0f) : 1.0f;
  const float cbias  = bias[col];
  float* cp = C + (size_t)(tn * 16 + hi * 8) * MCOLS + col;
  const float* rp = (MODE == 1) ? residual + (size_t)(tn * 16 + hi * 8) * DIM + col
                                : nullptr;
  #pragma unroll
  for (int rr = 0; rr < 8; ++rr) {
    float v = acc[rr] + cbias;
    if (MODE == 0) v *= cscale;
    else           v += rp[rr * DIM];
    cp[rr * MCOLS] = v;
  }
}

// ---- softmax stats init --------------------------------------------------
__global__ void k_zero_stats(unsigned* __restrict__ smax, float* __restrict__ den)
{
  int i = blockIdx.x * blockDim.x + threadIdx.x;
  smax[i] = 0u;         // smallest key, below the key of -inf
  den[i]  = 0.0f;
}

// ---- per-edge scores + fused segment max ---------------------------------
__global__ __launch_bounds__(256)
void k_scores(const float* __restrict__ qkv, const int* __restrict__ src,
              const int* __restrict__ dst, const float* __restrict__ attmask,
              float* __restrict__ score, unsigned* __restrict__ smax)
{
  int idx = blockIdx.x * blockDim.x + threadIdx.x;   // E*H threads
  int e = idx >> 3, h = idx & 7;
  int s = src[e], d = dst[e];
  const float4* kp = (const float4*)(qkv + (size_t)s * MQKV + DIM + h * HD_); // K block
  const float4* qp = (const float4*)(qkv + (size_t)d * MQKV + h * HD_);       // Q block
  float acc = 0.0f;
  #pragma unroll
  for (int j = 0; j < 8; ++j) {
    float4 a = kp[j], b = qp[j];
    acc += a.x * b.x + a.y * b.y + a.z * b.z + a.w * b.w;
  }
  bool m = (attmask[s] >= 0.0f) && (attmask[d] >= 0.0f);
  float sc = m ? acc : -10000.0f;
  score[idx] = sc;
  atomicMax(&smax[d * NH + h], f_orderable(sc));
}

// ---- exp(score - max) + segment sum --------------------------------------
__global__ void k_exp(float* __restrict__ score, const int* __restrict__ dst,
                      const unsigned* __restrict__ smax, float* __restrict__ den)
{
  int idx = blockIdx.x * blockDim.x + threadIdx.x;
  int e = idx >> 3, h = idx & 7;
  int d = dst[e];
  float mx = f_unorderable(smax[d * NH + h]);
  float es = expf(score[idx] - mx);
  score[idx] = es;
  atomicAdd(&den[d * NH + h], es);
}

// ---- normalize -----------------------------------------------------------
__global__ void k_norm(float* __restrict__ score, const int* __restrict__ dst,
                       const float* __restrict__ den)
{
  int idx = blockIdx.x * blockDim.x + threadIdx.x;
  int e = idx >> 3, h = idx & 7;
  score[idx] = score[idx] / den[dst[e] * NH + h];
}

// ---- diffusion init: h0 = exp(-1)*v ; gather = h0 ; hB = 0 ---------------
__global__ void k_diffuse_init(const float* __restrict__ qkv, float* __restrict__ hA,
                               float* __restrict__ gat, float* __restrict__ hB)
{
  int i = blockIdx.x * blockDim.x + threadIdx.x;     // N*DIM
  int n = i >> 8, c = i & 255;
  float v  = qkv[(size_t)n * MQKV + 2 * DIM + c];    // V block
  float h0 = 0.36787944117144233f * v;               // exp(-TEMP), TEMP=1
  hA[i]  = h0;
  gat[i] = h0;
  hB[i]  = 0.0f;
}

// ---- message passing: hout[dst] += attn(e,h) * hin[src] ------------------
__global__ __launch_bounds__(256)
void k_scatter(const float* __restrict__ attn, const int* __restrict__ src,
               const int* __restrict__ dst, const float* __restrict__ hin,
               float* __restrict__ hout)
{
  int idx = blockIdx.x * blockDim.x + threadIdx.x;   // E*H threads
  int e = idx >> 3, h = idx & 7;
  int s = src[e], d = dst[e];
  const float4* hs = (const float4*)(hin + (size_t)s * DIM + h * HD_);
  float* hd = hout + (size_t)d * DIM + h * HD_;
  __builtin_prefetch(hd, 1, 0);                      // warm the RMW target line
  float a = attn[idx];
  #pragma unroll
  for (int j = 0; j < 8; ++j) {
    float4 x = hs[j];
    atomicAdd(hd + 4 * j + 0, a * x.x);
    atomicAdd(hd + 4 * j + 1, a * x.y);
    atomicAdd(hd + 4 * j + 2, a * x.z);
    atomicAdd(hd + 4 * j + 3, a * x.w);
  }
}

// ---- gather += fact*hnew ; zero the buffer freed for the next round ------
__global__ void k_acc_zero(float* __restrict__ gat, const float* __restrict__ hnew,
                           float* __restrict__ hzero, float fact)
{
  int i = blockIdx.x * blockDim.x + threadIdx.x;     // N*DIM
  gat[i] += fact * hnew[i];
  hzero[i] = 0.0f;
}

// ---- LayerNorm: one wave32 per 256-float row -----------------------------
__global__ __launch_bounds__(256)
void k_layernorm(const float* __restrict__ y, const float* __restrict__ gamma,
                 const float* __restrict__ beta, float* __restrict__ out)
{
  int gw   = (blockIdx.x * blockDim.x + threadIdx.x) >> 5;   // row
  int lane = threadIdx.x & 31;
  const float4* row = (const float4*)(y + (size_t)gw * DIM);
  float4 a = row[lane];
  float4 b = row[lane + 32];
  float s = a.x + a.y + a.z + a.w + b.x + b.y + b.z + b.w;
  float q = a.x*a.x + a.y*a.y + a.z*a.z + a.w*a.w
          + b.x*b.x + b.y*b.y + b.z*b.z + b.w*b.w;
  #pragma unroll
  for (int m = 16; m >= 1; m >>= 1) {
    s += __shfl_xor(s, m, 32);
    q += __shfl_xor(q, m, 32);
  }
  float mu  = s * (1.0f / 256.0f);
  float var = q * (1.0f / 256.0f) - mu * mu;
  float rs  = rsqrtf(var + 1e-12f);
  int c0 = lane * 4, c1 = 128 + lane * 4;
  float4 o0, o1;
  o0.x = (a.x - mu) * rs * gamma[c0 + 0] + beta[c0 + 0];
  o0.y = (a.y - mu) * rs * gamma[c0 + 1] + beta[c0 + 1];
  o0.z = (a.z - mu) * rs * gamma[c0 + 2] + beta[c0 + 2];
  o0.w = (a.w - mu) * rs * gamma[c0 + 3] + beta[c0 + 3];
  o1.x = (b.x - mu) * rs * gamma[c1 + 0] + beta[c1 + 0];
  o1.y = (b.y - mu) * rs * gamma[c1 + 1] + beta[c1 + 1];
  o1.z = (b.z - mu) * rs * gamma[c1 + 2] + beta[c1 + 2];
  o1.w = (b.w - mu) * rs * gamma[c1 + 3] + beta[c1 + 3];
  float4* orow = (float4*)(out + (size_t)gw * DIM);
  orow[lane]      = o0;
  orow[lane + 32] = o1;
}

// ---- host launcher -------------------------------------------------------
extern "C" void kernel_launch(void* const* d_in, const int* in_sizes, int n_in,
                              void* d_out, int out_size, void* d_ws, size_t ws_size,
                              hipStream_t stream)
{
  const float* hidden  = (const float*)d_in[0];   // [N, D]
  const float* attmask = (const float*)d_in[1];   // [B, S] == [N]
  const int*   src     = (const int*)d_in[2];     // [E]
  const int*   dst     = (const int*)d_in[3];     // [E]
  const float* Wqkv    = (const float*)d_in[4];   // [D, 3D]
  const float* bqkv    = (const float*)d_in[5];   // [3D]
  const float* Wo      = (const float*)d_in[6];   // [D, D]
  const float* bo      = (const float*)d_in[7];   // [D]
  const float* gamma   = (const float*)d_in[8];   // [D]
  const float* beta    = (const float*)d_in[9];   // [D]
  float* out = (float*)d_out;

  // workspace carve-up (all fp32)
  float*    qkv   = (float*)d_ws;                          // N*768
  float*    score = qkv + (size_t)NNODE * MQKV;            // E*8 (scores -> es -> attn)
  unsigned* smax  = (unsigned*)(score + (size_t)NE * NH);  // N*8
  float*    den   = (float*)(smax + (size_t)NNODE * NH);   // N*8
  float*    hA    = den + (size_t)NNODE * NH;              // N*256
  float*    hB    = hA + (size_t)NNODE * DIM;              // N*256
  float*    gat   = hB + (size_t)NNODE * DIM;              // N*256
  float*    y     = gat + (size_t)NNODE * DIM;             // N*256

  const float qscale = 0.17677669529663687f;  // 1/sqrt(HD)

  // 1) QKV projection (+ bias, + q scaling): (N/16) * (768/128) = 6144 blocks
  k_wmma_gemm<0, MQKV><<<6144, 256, 0, stream>>>(hidden, Wqkv, bqkv, nullptr, qkv, qscale);

  // 2) softmax stats init
  k_zero_stats<<<(NNODE * NH) / 256, 256, 0, stream>>>(smax, den);

  // 3) edge scores + fused segment-max
  k_scores<<<(NE * NH) / 256, 256, 0, stream>>>(qkv, src, dst, attmask, score, smax);

  // 4) exp + segment-sum, 5) normalize
  k_exp<<<(NE * NH) / 256, 256, 0, stream>>>(score, dst, smax, den);
  k_norm<<<(NE * NH) / 256, 256, 0, stream>>>(score, dst, den);

  // 6) diffusion: h0 = exp(-1)*v, gather = h0
  k_diffuse_init<<<(NNODE * DIM) / 256, 256, 0, stream>>>(qkv, hA, gat, hB);

  // ki=1: fact=1
  k_scatter<<<(NE * NH) / 256, 256, 0, stream>>>(score, src, dst, hA, hB);
  k_acc_zero<<<(NNODE * DIM) / 256, 256, 0, stream>>>(gat, hB, hA, 1.0f);
  // ki=2: fact=1/2
  k_scatter<<<(NE * NH) / 256, 256, 0, stream>>>(score, src, dst, hB, hA);
  k_acc_zero<<<(NNODE * DIM) / 256, 256, 0, stream>>>(gat, hA, hB, 0.5f);
  // ki=3: fact=1/6
  k_scatter<<<(NE * NH) / 256, 256, 0, stream>>>(score, src, dst, hA, hB);
  k_acc_zero<<<(NNODE * DIM) / 256, 256, 0, stream>>>(gat, hB, hA, 1.0f / 6.0f);

  // 7) out projection + bias + residual: (N/16) * (256/128) = 2048 blocks
  k_wmma_gemm<1, DIM><<<2048, 256, 0, stream>>>(gat, Wo, bo, hidden, y, 1.0f);

  // 8) LayerNorm (one wave per row): 16384 rows / 8 waves = 2048 blocks
  k_layernorm<<<2048, 256, 0, stream>>>(y, gamma, beta, out);
}
